// LoRA_QKVlinear_61813169324213
// MI455X (gfx1250) — compile-verified, matrix operand order checked by reading, not run on hardware
//
#include <hip/hip_runtime.h>
#include <hip/hip_bf16.h>

typedef __attribute__((ext_vector_type(16))) __bf16 v16bf;
typedef __attribute__((ext_vector_type(8)))  __bf16 v8bf;
typedef __attribute__((ext_vector_type(8)))  float  v8f;
typedef __attribute__((ext_vector_type(4)))  float  v4f;

constexpr int I_DIM = 1024;   // inner dim
constexpr int O_DIM = 1024;   // per-projection out dim
constexpr int N_DIM = 3072;   // 3*O
constexpr int M_DIM = 16384;  // B*S
constexpr int R_DIM = 256;    // vera rank

// ---------------------------------------------------------------------------
// CDNA5 async copy: global -> LDS, tracked by ASYNCcnt (no VGPR staging).
// GV addressing mode: 64-bit VGPR address, LDS dest address in VDST VGPR.
// ---------------------------------------------------------------------------
__device__ __forceinline__ void async_ld_b128(unsigned lds_addr, const void* g) {
  asm volatile("global_load_async_to_lds_b128 %0, %1, off"
               :: "v"(lds_addr), "v"((unsigned long long)(uintptr_t)g)
               : "memory");
}

template <int N>
__device__ __forceinline__ void wait_async() {
#if __has_builtin(__builtin_amdgcn_s_wait_asynccnt)
  __builtin_amdgcn_s_wait_asynccnt(N);
#else
  asm volatile("s_wait_asynccnt %0" :: "n"(N) : "memory");
#endif
}

__device__ __forceinline__ unsigned lds_off(const void* p) {
  // generic shared-space pointer: addr[31:0] is the LDS byte offset (ISA 10.2)
  return (unsigned)(uintptr_t)p;
}

// ---------------------------------------------------------------------------
// Stage 0: f32 -> bf16 bulk convert (used for x and the Q rows of W)
// ---------------------------------------------------------------------------
__global__ void cvt_f32_to_bf16(const float* __restrict__ src,
                                __bf16* __restrict__ dst, int n8) {
  int i = blockIdx.x * blockDim.x + threadIdx.x;
  if (i >= n8) return;
  const v4f* s4 = (const v4f*)src;
  v4f a = s4[2 * i + 0];
  v4f b = s4[2 * i + 1];
  v8bf o;
  o[0] = (__bf16)a[0]; o[1] = (__bf16)a[1];
  o[2] = (__bf16)a[2]; o[3] = (__bf16)a[3];
  o[4] = (__bf16)b[0]; o[5] = (__bf16)b[1];
  o[6] = (__bf16)b[2]; o[7] = (__bf16)b[3];
  ((v8bf*)dst)[i] = o;
}

// ---------------------------------------------------------------------------
// Stage 1: fused KV weight rows:  W[1024+row] = bf16(base + b_k[o] * (s @ A))
//          where s[r] = B[o,r] * d_k[r].   One block (256 thr) per KV row.
// ---------------------------------------------------------------------------
__global__ __launch_bounds__(256)
void kv_delta_rows(const float* __restrict__ base_w,
                   const float* __restrict__ vA,   // (R, I)
                   const float* __restrict__ vB,   // (O, R)
                   const float* __restrict__ vd,   // (2, R)
                   const float* __restrict__ vb,   // (2, O)
                   __bf16* __restrict__ Wb) {
  __shared__ float s[R_DIM];
  const int row = blockIdx.x;            // 0..2047 (K then V)
  const int k = row >> 10;
  const int o = row & (O_DIM - 1);
  const int t = threadIdx.x;

  s[t] = vB[o * R_DIM + t] * vd[k * R_DIM + t];
  const float bs = vb[k * O_DIM + o];    // SCALE == 1.0 folded in
  __syncthreads();

  const float*  wrow = base_w + (size_t)(O_DIM + row) * I_DIM;
  __bf16*       orow = Wb     + (size_t)(O_DIM + row) * I_DIM;

  #pragma unroll
  for (int ii = 0; ii < 4; ++ii) {
    const int i = t + ii * 256;
    float acc = 0.f;
    #pragma unroll 4
    for (int r = 0; r < R_DIM; ++r) acc += s[r] * vA[r * I_DIM + i];
    orow[i] = (__bf16)(wrow[i] + bs * acc);
  }
}

// ---------------------------------------------------------------------------
// Stage 2: bf16 WMMA GEMM:  out(M,3072) = Xb(M,1024) @ Wb^T + bias
// Block tile 128x256, 8 waves (wave32) of 64x64 each -> 16 wmma per K-step.
// Double-buffered LDS filled by global_load_async_to_lds_b128 (ASYNCcnt).
// Steady-state loop is branch-free; final K-step peeled.
// ---------------------------------------------------------------------------
constexpr int BM  = 128;
constexpr int BN  = 256;
constexpr int BK  = 32;
constexpr int LDP = 40;                      // LDS pitch in halfs (80 B rows)
constexpr unsigned XBUF = BM * LDP * 2;      // bytes per X buffer (10 KB)
constexpr unsigned WBUF = BN * LDP * 2;      // bytes per W buffer (20 KB)

__global__ __launch_bounds__(256)
void gemm_wmma_bf16(const __bf16* __restrict__ Xb,
                    const __bf16* __restrict__ Wb,
                    const float*  __restrict__ bias,
                    float* __restrict__ out) {
  __shared__ __attribute__((aligned(32))) __bf16 xs[2 * BM * LDP];  // 20 KB
  __shared__ __attribute__((aligned(32))) __bf16 wl[2 * BN * LDP];  // 40 KB

  const int t    = threadIdx.x;
  const int wave = t >> 5;
  const int lane = t & 31;
  const int lrow = lane & 15;
  const int lhi  = lane >> 4;

  const int m0 = blockIdx.y * BM;
  const int n0 = blockIdx.x * BN;
  const int wm = (wave & 1) * 64;   // 2 waves along M
  const int wn = (wave >> 1) * 64;  // 4 waves along N

  v8f acc[4][4];
  #pragma unroll
  for (int mt = 0; mt < 4; ++mt)
    #pragma unroll
    for (int nt = 0; nt < 4; ++nt) {
      v8f z = {};
      acc[mt][nt] = z;
    }

  // per-thread async copy maps (16 B per transfer)
  const int xr = t >> 1, xc = (t & 1) * 16;               // X: 2 thr/row
  const __bf16* xsrc = Xb + (size_t)(m0 + xr) * I_DIM + xc;
  const __bf16* wsrc = Wb + (size_t)(n0 + t) * I_DIM;
  const unsigned xdst = lds_off(xs + xr * LDP + xc);
  const unsigned wdst = lds_off(wl + t * LDP);

  auto stage = [&](int buf, int kk) {
    const unsigned xo = xdst + (unsigned)buf * XBUF;
    async_ld_b128(xo,      xsrc + kk);
    async_ld_b128(xo + 16, xsrc + kk + 8);
    const unsigned wo = wdst + (unsigned)buf * WBUF;
    async_ld_b128(wo,      wsrc + kk);
    async_ld_b128(wo + 16, wsrc + kk + 8);
    async_ld_b128(wo + 32, wsrc + kk + 16);
    async_ld_b128(wo + 48, wsrc + kk + 24);
  };

  auto compute = [&](int buf) {
    const __bf16* xcur = xs + buf * (BM * LDP);
    const __bf16* wcur = wl + buf * (BN * LDP);

    v16bf afrag[4], bfrag[4];
    #pragma unroll
    for (int mt = 0; mt < 4; ++mt) {
      const __bf16* p = xcur + (wm + mt * 16 + lrow) * LDP + lhi * 16;
      union { v16bf v; v8bf h[2]; } u;
      u.h[0] = *(const v8bf*)p;
      u.h[1] = *(const v8bf*)(p + 8);
      afrag[mt] = u.v;
    }
    #pragma unroll
    for (int nt = 0; nt < 4; ++nt) {
      const __bf16* p = wcur + (wn + nt * 16 + lrow) * LDP + lhi * 16;
      union { v16bf v; v8bf h[2]; } u;
      u.h[0] = *(const v8bf*)p;
      u.h[1] = *(const v8bf*)(p + 8);
      bfrag[nt] = u.v;
    }

    #pragma unroll
    for (int mt = 0; mt < 4; ++mt)
      #pragma unroll
      for (int nt = 0; nt < 4; ++nt)
        acc[mt][nt] = __builtin_amdgcn_wmma_f32_16x16x32_bf16(
            false, afrag[mt], false, bfrag[nt],
            (short)0, acc[mt][nt], false, false);
  };

  stage(0, 0);                      // prologue fill of buffer 0
  int cur = 0;

  // steady state: branch-free (final K-step peeled)
  for (int kk = 0; kk < I_DIM - BK; kk += BK) {
    stage(cur ^ 1, kk + BK);        // kick off next tile's DMA first
    wait_async<6>();                // drain all but the 6 just issued
    __syncthreads();                // all waves' DMA for `cur` landed
    compute(cur);
    __syncthreads();                // safe to overwrite `cur` next round
    cur ^= 1;
  }

  // peeled final K-step
  wait_async<0>();
  __syncthreads();
  compute(cur);

  // epilogue: bias fused; 16-lane groups write 16 consecutive floats
  #pragma unroll
  for (int nt = 0; nt < 4; ++nt) {
    const int col = n0 + wn + nt * 16 + lrow;
    const float bv = bias[col];
    #pragma unroll
    for (int mt = 0; mt < 4; ++mt) {
      #pragma unroll
      for (int v = 0; v < 8; ++v) {
        const int row = m0 + wm + mt * 16 + lhi * 8 + v;
        out[(size_t)row * N_DIM + col] = acc[mt][nt][v] + bv;
      }
    }
  }
}

// ---------------------------------------------------------------------------
extern "C" void kernel_launch(void* const* d_in, const int* in_sizes, int n_in,
                              void* d_out, int out_size, void* d_ws, size_t ws_size,
                              hipStream_t stream) {
  const float* x  = (const float*)d_in[0];  // (4,4096,1024)
  const float* bw = (const float*)d_in[1];  // (3072,1024)
  const float* bb = (const float*)d_in[2];  // (3072,)
  const float* vA = (const float*)d_in[3];  // (256,1024)
  const float* vB = (const float*)d_in[4];  // (1024,256)
  const float* vd = (const float*)d_in[5];  // (2,256)
  const float* vb = (const float*)d_in[6];  // (2,1024)
  float* out = (float*)d_out;

  // workspace: Wb (3072*1024 bf16 = 6 MB) | Xb (16384*1024 bf16 = 32 MB)
  __bf16* Wb = (__bf16*)d_ws;
  __bf16* Xb = (__bf16*)((char*)d_ws + (size_t)N_DIM * I_DIM * sizeof(__bf16));

  // x -> bf16
  {
    int n8 = (M_DIM * I_DIM) / 8;
    cvt_f32_to_bf16<<<(n8 + 255) / 256, 256, 0, stream>>>(x, Xb, n8);
  }
  // Q rows of W (delta == 0) -> bf16
  {
    int n8 = (O_DIM * I_DIM) / 8;
    cvt_f32_to_bf16<<<(n8 + 255) / 256, 256, 0, stream>>>(bw, Wb, n8);
  }
  // K,V rows fused with VERA delta
  kv_delta_rows<<<2 * O_DIM, 256, 0, stream>>>(bw, vA, vB, vd, vb, Wb);

  // main GEMM: grid (N/256, M/128) = (12, 128)
  gemm_wmma_bf16<<<dim3(N_DIM / BN, M_DIM / BM), 256, 0, stream>>>(Xb, Wb, bb, out);
}
